// UniCrystalFormer_59940563583893
// MI455X (gfx1250) — matrix operand order repaired and checked by reading, not compiled, not run on hardware
//
#include <hip/hip_runtime.h>
#include <math.h>

#define NN 4096
#define EEDGE 32768
#define BGRAPH 32

typedef __attribute__((ext_vector_type(16))) _Float16 v16h;
typedef __attribute__((ext_vector_type(8)))  _Float16 v8h;
typedef __attribute__((ext_vector_type(4)))  _Float16 v4h;
typedef __attribute__((ext_vector_type(8)))  float    v8f;

__device__ __forceinline__ float dsig(float x){ return 1.f/(1.f+__expf(-x)); }

// B fragment: weights pre-transposed to [Nc][K] f16, L2-resident -> direct global loads.
__device__ __forceinline__ v16h loadB(const _Float16* __restrict__ Wt, int K,
                                      int row, int k0, int half){
  const _Float16* p = Wt + (size_t)row*K + k0 + 8*half;
  v8h lo = *(const v8h*)p;
  v8h hi = *(const v8h*)(p + 16);
  return __builtin_shufflevector(lo, hi, 0,1,2,3,4,5,6,7,8,9,10,11,12,13,14,15);
}

// ====================== WMMA GEMM, f32 activations ======================
// Y = act( X[M,K] @ W + bias ); W given transposed f16 [Nc][K]. K % 32 == 0.
// block=128thr=4 waves; tile 32(M)x64(N); each wave: 16-wide N subtile, 2 WMMAs/step.
template<int ACT, bool ACCUM, bool YH>
__global__ __launch_bounds__(128) void gemm_f32in(
    const float* __restrict__ X, const _Float16* __restrict__ Wt,
    const float* __restrict__ bias, void* __restrict__ Yv,
    int ldy, int M, int K, int Nc)
{
  __shared__ _Float16 As[32*32];
  const int tid  = threadIdx.x;
  const int wv   = tid >> 5;
  const int lane = tid & 31;
  const int half = lane >> 4;
  const int lm   = lane & 15;
  const long m0  = (long)blockIdx.y * 32;
  const int  n0  = blockIdx.x * 64;
  const int  nr  = n0 + (wv<<4) + lm;
  v8f acc0 = {0.f,0.f,0.f,0.f,0.f,0.f,0.f,0.f};
  v8f acc1 = {0.f,0.f,0.f,0.f,0.f,0.f,0.f,0.f};

  for (int k0 = 0; k0 < K; k0 += 32) {
    // stage A 32x32: float4 load + cvt + 8B LDS store, x2 per thread
#pragma unroll
    for (int t = 0; t < 2; ++t) {
      int idx = (tid + t*128) * 4;
      int mm = idx >> 5, kk = idx & 31;
      const float4 f = *(const float4*)(X + (m0+mm)*(long)K + k0 + kk);
      v4h h4; h4[0]=(_Float16)f.x; h4[1]=(_Float16)f.y; h4[2]=(_Float16)f.z; h4[3]=(_Float16)f.w;
      *(v4h*)&As[mm*32 + kk] = h4;
    }
    if (k0 + 32 < K) __builtin_prefetch(X + (m0 + (tid >> 3))*(long)K + k0 + 32, 0, 1);
    __syncthreads();
    v8h a0lo = *(const v8h*)&As[lm*32 + 8*half];
    v8h a0hi = *(const v8h*)&As[lm*32 + 16 + 8*half];
    v8h a1lo = *(const v8h*)&As[(16+lm)*32 + 8*half];
    v8h a1hi = *(const v8h*)&As[(16+lm)*32 + 16 + 8*half];
    v16h a0 = __builtin_shufflevector(a0lo, a0hi, 0,1,2,3,4,5,6,7,8,9,10,11,12,13,14,15);
    v16h a1 = __builtin_shufflevector(a1lo, a1hi, 0,1,2,3,4,5,6,7,8,9,10,11,12,13,14,15);
    v16h b  = loadB(Wt, K, nr, k0, half);
    acc0 = __builtin_amdgcn_wmma_f32_16x16x32_f16(false, a0, false, b, (short)0, acc0, false, false);
    acc1 = __builtin_amdgcn_wmma_f32_16x16x32_f16(false, a1, false, b, (short)0, acc1, false, false);
    __syncthreads();
  }

  const float bv = bias ? bias[nr] : 0.f;
#pragma unroll
  for (int r = 0; r < 8; ++r) {
    long mA = m0 + r + 8*half, mB = mA + 16;
    float v0 = acc0[r] + bv, v1 = acc1[r] + bv;
    if (ACT == 1) { v0 = v0*dsig(v0); v1 = v1*dsig(v1); }
    else if (ACT == 2) { v0 = dsig(v0); v1 = dsig(v1); }
    long i0 = mA*(long)ldy + nr, i1 = mB*(long)ldy + nr;
    if (YH) {
      _Float16* Y = (_Float16*)Yv; Y[i0] = (_Float16)v0; Y[i1] = (_Float16)v1;
    } else {
      float* Y = (float*)Yv;
      if (ACCUM) { Y[i0] += v0; Y[i1] += v1; } else { Y[i0] = v0; Y[i1] = v1; }
    }
  }
}

// ====================== WMMA GEMM, f16 activations ======================
template<int ACT, bool YH>
__global__ __launch_bounds__(128) void gemm_f16in(
    const _Float16* __restrict__ Xh, const _Float16* __restrict__ Wt,
    const float* __restrict__ bias, void* __restrict__ Yv,
    int ldy, int M, int K, int Nc)
{
  __shared__ _Float16 As[32*32];
  const int tid  = threadIdx.x;
  const int wv   = tid >> 5;
  const int lane = tid & 31;
  const int half = lane >> 4;
  const int lm   = lane & 15;
  const long m0  = (long)blockIdx.y * 32;
  const int  n0  = blockIdx.x * 64;
  const int  nr  = n0 + (wv<<4) + lm;
  v8f acc0 = {0.f,0.f,0.f,0.f,0.f,0.f,0.f,0.f};
  v8f acc1 = {0.f,0.f,0.f,0.f,0.f,0.f,0.f,0.f};

  for (int k0 = 0; k0 < K; k0 += 32) {
    // stage A 32x32 halves: one 16B copy per thread (4 threads per row)
    {
      int mm = tid >> 2, off = (tid & 3) * 8;
      *(v8h*)&As[mm*32 + off] = *(const v8h*)(Xh + (m0+mm)*(long)K + k0 + off);
    }
    if (k0 + 32 < K) __builtin_prefetch(Xh + (m0 + (tid >> 2))*(long)K + k0 + 32, 0, 1);
    __syncthreads();
    v8h a0lo = *(const v8h*)&As[lm*32 + 8*half];
    v8h a0hi = *(const v8h*)&As[lm*32 + 16 + 8*half];
    v8h a1lo = *(const v8h*)&As[(16+lm)*32 + 8*half];
    v8h a1hi = *(const v8h*)&As[(16+lm)*32 + 16 + 8*half];
    v16h a0 = __builtin_shufflevector(a0lo, a0hi, 0,1,2,3,4,5,6,7,8,9,10,11,12,13,14,15);
    v16h a1 = __builtin_shufflevector(a1lo, a1hi, 0,1,2,3,4,5,6,7,8,9,10,11,12,13,14,15);
    v16h b  = loadB(Wt, K, nr, k0, half);
    acc0 = __builtin_amdgcn_wmma_f32_16x16x32_f16(false, a0, false, b, (short)0, acc0, false, false);
    acc1 = __builtin_amdgcn_wmma_f32_16x16x32_f16(false, a1, false, b, (short)0, acc1, false, false);
    __syncthreads();
  }

  const float bv = bias ? bias[nr] : 0.f;
#pragma unroll
  for (int r = 0; r < 8; ++r) {
    long mA = m0 + r + 8*half, mB = mA + 16;
    float v0 = acc0[r] + bv, v1 = acc1[r] + bv;
    if (ACT == 1) { v0 = v0*dsig(v0); v1 = v1*dsig(v1); }
    else if (ACT == 2) { v0 = dsig(v0); v1 = dsig(v1); }
    long i0 = mA*(long)ldy + nr, i1 = mB*(long)ldy + nr;
    if (YH) { _Float16* Y = (_Float16*)Yv; Y[i0] = (_Float16)v0; Y[i1] = (_Float16)v1; }
    else    { float* Y = (float*)Yv; Y[i0] = v0; Y[i1] = v1; }
  }
}

static void gemm32(hipStream_t st, const float* X, const _Float16* Wt, const float* bias,
                   void* Y, int ldy, int M, int K, int Nc, int act, bool accum, bool yh)
{
  dim3 g((unsigned)(Nc/64), (unsigned)(M/32)), b(128);
  if (accum)      gemm_f32in<0,true ,false><<<g,b,0,st>>>(X,Wt,bias,Y,ldy,M,K,Nc);
  else if (yh)    gemm_f32in<1,false,true ><<<g,b,0,st>>>(X,Wt,bias,Y,ldy,M,K,Nc); // silu->f16
  else if (act==0)gemm_f32in<0,false,false><<<g,b,0,st>>>(X,Wt,bias,Y,ldy,M,K,Nc);
  else if (act==1)gemm_f32in<1,false,false><<<g,b,0,st>>>(X,Wt,bias,Y,ldy,M,K,Nc);
  else            gemm_f32in<2,false,false><<<g,b,0,st>>>(X,Wt,bias,Y,ldy,M,K,Nc);
}
static void gemm16(hipStream_t st, const _Float16* Xh, const _Float16* Wt, const float* bias,
                   void* Y, int ldy, int M, int K, int Nc, int act, bool yh)
{
  dim3 g((unsigned)(Nc/64), (unsigned)(M/32)), b(128);
  if (yh) gemm_f16in<1,true ><<<g,b,0,st>>>(Xh,Wt,bias,Y,ldy,M,K,Nc);  // silu -> f16
  else if (act==1) gemm_f16in<1,false><<<g,b,0,st>>>(Xh,Wt,bias,Y,ldy,M,K,Nc);
  else    gemm_f16in<0,false><<<g,b,0,st>>>(Xh,Wt,bias,Y,ldy,M,K,Nc);
}

// ====================== weight transpose/convert ======================
__global__ void k_wt(const float* w, _Float16* wt, int K, int Nc, int Kpad, size_t n){
  size_t i = blockIdx.x*(size_t)256 + threadIdx.x; if (i >= n) return;
  int nr = (int)(i / Kpad); int k = (int)(i % Kpad);
  wt[i] = (k < K) ? (_Float16)w[(size_t)k*Nc + nr] : (_Float16)0.f;
}
__global__ void k_padmeg(const float* in, float* out, size_t n){
  size_t i = blockIdx.x*(size_t)256 + threadIdx.x; if (i >= n) return;
  size_t node = i >> 5; int j = (int)(i & 31);
  out[i] = (j < 16) ? in[node*16 + j] : 0.f;
}

// ====================== elementwise kernels ======================
__global__ void k_fill(float* p, float v, size_t n){
  size_t i = blockIdx.x*(size_t)256 + threadIdx.x; if (i < n) p[i] = v;
}
__global__ void k_add(const float* a, const float* b, float* y, size_t n){
  size_t i = blockIdx.x*(size_t)256 + threadIdx.x; if (i < n) y[i] = a[i] + b[i];
}
__global__ void k_emb(const int* an, const float* tab, float* cat, size_t n){
  size_t i = blockIdx.x*(size_t)256 + threadIdx.x; if (i >= n) return;
  size_t node = i >> 7; int f = (int)(i & 127);
  cat[node*256 + f] = tab[(size_t)an[node]*128 + f];
}
__global__ void k_fuse(const float* cat, const float* g, float* out, size_t n){
  size_t i = blockIdx.x*(size_t)256 + threadIdx.x; if (i >= n) return;
  size_t node = i >> 7; int f = (int)(i & 127);
  float emb = cat[node*256 + f], meg = cat[node*256 + 128 + f], gg = g[i];
  out[i] = gg*emb + (1.f-gg)*meg;
}
__global__ void k_dist(const float* ev, float* d, int n){
  int i = blockIdx.x*256 + threadIdx.x; if (i >= n) return;
  float a = ev[3*i], b = ev[3*i+1], c = ev[3*i+2];
  d[i] = sqrtf(a*a + b*b + c*c);
}
__global__ void k_rbf(const float* d, float* o, size_t n){
  size_t i = blockIdx.x*(size_t)256 + threadIdx.x; if (i >= n) return;
  size_t e = i >> 7; int bin = (int)(i & 127);
  float ctr = (float)bin * (8.f/127.f);
  float gamma = (127.f/8.f)*(127.f/8.f);
  float dt = d[e] - ctr;
  o[i] = __expf(-gamma*dt*dt);
}
__global__ void k_cart_cat(const float* h, const float* e, const int* src, const int* dst,
                           _Float16* c, size_t n){
  size_t i = blockIdx.x*(size_t)256 + threadIdx.x; if (i >= n) return;
  size_t ed = i / 384; int j = (int)(i % 384);
  float v;
  if (j < 128)      v = h[(size_t)dst[ed]*128 + j];
  else if (j < 256) v = h[(size_t)src[ed]*128 + (j-128)];
  else              v = e[ed*128 + (j-256)];
  c[i] = (_Float16)v;
}
__global__ void k_cart_scatter(const float* gate, const float* msg, const float* dist,
                               const int* dst, float* agg, size_t n){
  size_t i = blockIdx.x*(size_t)256 + threadIdx.x; if (i >= n) return;
  size_t ed = i >> 7; int f = (int)(i & 127);
  float d = dist[ed];
  float env = (d <= 8.f) ? 0.5f*(cosf(3.14159265358979f*d*0.125f) + 1.f) : 0.f;
  float s = dsig(gate[i]) * env;
  atomicAdd(&agg[(size_t)dst[ed]*128 + f], s * msg[i]);
}
__global__ void k_kcat_alpha(const float* kb, const float* qb, const float* ee,
                             const int* src, const int* dst, int hh, float* alpha, size_t n){
  size_t i = blockIdx.x*(size_t)256 + threadIdx.x; if (i >= n) return;
  size_t ed = i / 384; int j = (int)(i % 384); int f = j & 127;
  float v;
  if (j < 128)      v = kb[(size_t)dst[ed]*512 + hh*128 + f];
  else if (j < 256) v = kb[(size_t)src[ed]*512 + hh*128 + f];
  else              v = ee[ed*512 + hh*128 + f];
  float qv = qb[(size_t)dst[ed]*512 + hh*128 + f];
  alpha[i] = v * qv * 0.05103103630798288f; // 1/sqrt(3*128)
}
__global__ void k_vcat(const float* vb, const float* ee, const int* src, const int* dst,
                       int hh, _Float16* out, size_t n){
  size_t i = blockIdx.x*(size_t)256 + threadIdx.x; if (i >= n) return;
  size_t ed = i / 384; int j = (int)(i % 384); int f = j & 127;
  float v;
  if (j < 128)      v = vb[(size_t)dst[ed]*512 + hh*128 + f];
  else if (j < 256) v = vb[(size_t)src[ed]*512 + hh*128 + f];
  else              v = ee[ed*512 + hh*128 + f];
  out[i] = (_Float16)v;
}
__global__ void k_mat_scatter(const float* msg, const int* dst, float* out, size_t n){
  size_t i = blockIdx.x*(size_t)256 + threadIdx.x; if (i >= n) return;
  size_t ed = i >> 7; int f = (int)(i & 127);
  atomicAdd(&out[(size_t)dst[ed]*128 + f], 0.25f * msg[i]);
}
__global__ void k_lstm(const float* z, float* h, float* c, size_t n){
  size_t i = blockIdx.x*(size_t)256 + threadIdx.x; if (i >= n) return;
  size_t g = i >> 7; int f = (int)(i & 127);
  float zi = z[g*512 + f], zf = z[g*512 + 128 + f];
  float zg = z[g*512 + 256 + f], zo = z[g*512 + 384 + f];
  float cc = dsig(zf)*c[i] + dsig(zi)*tanhf(zg);
  c[i] = cc;
  h[i] = dsig(zo)*tanhf(cc);
}
__global__ void k_expa(const float* eng, const int* batch, const float* mB, float* a, int n){
  int i = blockIdx.x*256 + threadIdx.x; if (i >= n) return;
  a[i] = __expf(eng[i] - mB[batch[i]]);
}
__global__ void k_qstar(const float* h, const float* r, float* q, size_t n){
  size_t i = blockIdx.x*(size_t)256 + threadIdx.x; if (i >= n) return;
  size_t g = i >> 8; int j = (int)(i & 255);
  q[i] = (j < 128) ? h[g*128 + j] : r[g*128 + (j-128)];
}
__global__ void k_gn_accum(const float* x, const int* batch, float* gsum, float* gcnt, size_t n){
  size_t i = blockIdx.x*(size_t)256 + threadIdx.x; if (i >= n) return;
  size_t node = i >> 7; int f = (int)(i & 127); int g = batch[node];
  atomicAdd(&gsum[(size_t)g*128 + f], x[i]);
  if (f == 0) atomicAdd(&gcnt[g], 1.f);
}
__global__ void k_gn_center(const float* x, const int* batch, const float* gsum,
                            const float* gcnt, const float* alpha, float* xc, float* gvar, size_t n){
  size_t i = blockIdx.x*(size_t)256 + threadIdx.x; if (i >= n) return;
  size_t node = i >> 7; int f = (int)(i & 127); int g = batch[node];
  float cnt = fmaxf(gcnt[g], 1.f);
  float m = gsum[(size_t)g*128 + f] / cnt;
  float v = x[i] - alpha[f]*m;
  xc[i] = v;
  atomicAdd(&gvar[(size_t)g*128 + f], v*v);
}
__global__ void k_gn_final(const float* xc, const int* batch, const float* gvar,
                           const float* gcnt, const float* gg, const float* bb,
                           const float* base, float* y, size_t n){
  size_t i = blockIdx.x*(size_t)256 + threadIdx.x; if (i >= n) return;
  size_t node = i >> 7; int f = (int)(i & 127); int g = batch[node];
  float cnt = fmaxf(gcnt[g], 1.f);
  float var = gvar[(size_t)g*128 + f] / cnt;
  y[i] = base[i] + gg[f]*xc[i]*rsqrtf(var + 1e-5f) + bb[f];
}

// ====================== block-reduce kernels ======================
__global__ __launch_bounds__(128) void k_ln(float* x, const float* g, const float* b, int dim){
  __shared__ float r1[128], r2[128];
  const int tid = threadIdx.x;
  float* xr = x + (size_t)blockIdx.x * dim;
  float s = 0.f, s2 = 0.f;
  for (int j = tid; j < dim; j += 128){ float v = xr[j]; s += v; s2 += v*v; }
  r1[tid] = s; r2[tid] = s2; __syncthreads();
  for (int o = 64; o; o >>= 1){ if (tid < o){ r1[tid]+=r1[tid+o]; r2[tid]+=r2[tid+o]; } __syncthreads(); }
  float m = r1[0]/dim, var = r2[0]/dim - m*m, rs = rsqrtf(var + 1e-5f);
  for (int j = tid; j < dim; j += 128) xr[j] = g[j]*(xr[j]-m)*rs + b[j];
}
// out_h = (f16)( msg * sigmoid(LayerNorm(alpha)) ), rows of dim 384
__global__ __launch_bounds__(128) void k_ln_sigmul_h(const float* alpha, const float* g,
                                                     const float* b, const float* msg,
                                                     _Float16* out){
  __shared__ float r1[128], r2[128];
  const int tid = threadIdx.x;
  const float* xr = alpha + (size_t)blockIdx.x * 384;
  float s = 0.f, s2 = 0.f;
  for (int j = tid; j < 384; j += 128){ float v = xr[j]; s += v; s2 += v*v; }
  r1[tid] = s; r2[tid] = s2; __syncthreads();
  for (int o = 64; o; o >>= 1){ if (tid < o){ r1[tid]+=r1[tid+o]; r2[tid]+=r2[tid+o]; } __syncthreads(); }
  float m = r1[0]/384.f, var = r2[0]/384.f - m*m, rs = rsqrtf(var + 1e-5f);
  for (int j = tid; j < 384; j += 128){
    float v = g[j]*(xr[j]-m)*rs + b[j];
    size_t idx = (size_t)blockIdx.x*384 + j;
    out[idx] = (_Float16)(msg[idx] * dsig(v));
  }
}
__global__ __launch_bounds__(256) void k_colstats(const float* x, int M, float* mean, float* rstd){
  __shared__ float r1[256], r2[256];
  const int f = blockIdx.x, tid = threadIdx.x;
  float s = 0.f, s2 = 0.f;
  for (int r = tid; r < M; r += 256){ float v = x[(size_t)r*128 + f]; s += v; s2 += v*v; }
  r1[tid] = s; r2[tid] = s2; __syncthreads();
  for (int o = 128; o; o >>= 1){ if (tid < o){ r1[tid]+=r1[tid+o]; r2[tid]+=r2[tid+o]; } __syncthreads(); }
  if (tid == 0){ float m = r1[0]/M; mean[f] = m; rstd[f] = rsqrtf(r2[0]/M - m*m + 1e-5f); }
}
template<bool RESID>
__global__ void k_bn_silu(const float* x, const float* mean, const float* rstd,
                          const float* g, const float* b, const float* base, float* y, size_t n){
  size_t i = blockIdx.x*(size_t)256 + threadIdx.x; if (i >= n) return;
  int f = (int)(i & 127);
  float v = g[f]*(x[i]-mean[f])*rstd[f] + b[f];
  v = v * dsig(v);
  y[i] = RESID ? base[i] + v : v;
}
__global__ __launch_bounds__(128) void k_beta(const float* out, const float* xr,
                                              const float* bw, float* y){
  __shared__ float red[128];
  const int node = blockIdx.x, t = threadIdx.x;
  float o = out[(size_t)node*128 + t], x = xr[(size_t)node*128 + t];
  red[t] = o*bw[t] + x*bw[128+t] + (o-x)*bw[256+t];
  __syncthreads();
  for (int s = 64; s; s >>= 1){ if (t < s) red[t] += red[t+s]; __syncthreads(); }
  float beta = dsig(red[0]);
  y[(size_t)node*128 + t] = beta*x + (1.f-beta)*o;
}
__global__ __launch_bounds__(128) void k_eng(const float* x, const float* h,
                                             const int* batch, float* eng){
  __shared__ float red[128];
  const int node = blockIdx.x, t = threadIdx.x;
  red[t] = x[(size_t)node*128 + t] * h[(size_t)batch[node]*128 + t];
  __syncthreads();
  for (int s = 64; s; s >>= 1){ if (t < s) red[t] += red[t+s]; __syncthreads(); }
  if (t == 0) eng[node] = red[0];
}
__global__ __launch_bounds__(256) void k_segmax(const float* eng, const int* batch, float* mB, int n){
  __shared__ float red[256];
  const int g = blockIdx.x, t = threadIdx.x;
  float m = -3e38f;
  for (int i = t; i < n; i += 256) if (batch[i] == g) m = fmaxf(m, eng[i]);
  red[t] = m; __syncthreads();
  for (int s = 128; s; s >>= 1){ if (t < s) red[t] = fmaxf(red[t], red[t+s]); __syncthreads(); }
  if (t == 0) mB[g] = red[0];
}
__global__ __launch_bounds__(256) void k_segsum1(const float* a, const int* batch, float* sB, int n){
  __shared__ float red[256];
  const int g = blockIdx.x, t = threadIdx.x;
  float s = 0.f;
  for (int i = t; i < n; i += 256) if (batch[i] == g) s += a[i];
  red[t] = s; __syncthreads();
  for (int o = 128; o; o >>= 1){ if (t < o) red[t] += red[t+o]; __syncthreads(); }
  if (t == 0) sB[g] = red[0];
}
__global__ __launch_bounds__(128) void k_rgather(const float* a, const float* x,
                                                 const int* batch, const float* sB,
                                                 float* r, int n){
  const int g = blockIdx.x, f = threadIdx.x;
  float inv = 1.f/(sB[g] + 1e-16f);
  float acc = 0.f;
  for (int i = 0; i < n; ++i) if (batch[i] == g) acc += a[i]*inv * x[(size_t)i*128 + f];
  r[(size_t)g*128 + f] = acc;
}
__global__ __launch_bounds__(128) void k_fc2k(const float* x, const float* w,
                                              const float* b, float* out){
  __shared__ float red[128];
  const int g = blockIdx.x, t = threadIdx.x;
  red[t] = x[(size_t)g*128 + t] * w[t];
  __syncthreads();
  for (int s = 64; s; s >>= 1){ if (t < s) red[t] += red[t+s]; __syncthreads(); }
  if (t == 0) out[g] = red[0] + b[0];
}

// ====================== host ======================
struct LinP { const float *b, *w; };

static inline unsigned nb(size_t n){ return (unsigned)((n + 255) / 256); }

extern "C" void kernel_launch(void* const* d_in, const int* in_sizes, int n_in,
                              void* d_out, int out_size, void* d_ws, size_t ws_size,
                              hipStream_t stream)
{
  (void)in_sizes; (void)n_in; (void)out_size; (void)ws_size;
  const int*   an     = (const int*)d_in[0];
  const float* meg_in = (const float*)d_in[1];
  const float* evec   = (const float*)d_in[2];
  const int*   eidx   = (const int*)d_in[3];
  const float* cdist  = (const float*)d_in[4];
  const int*   batch  = (const int*)d_in[5];
  const float* P      = (const float*)d_in[6];
  const int* src = eidx;
  const int* dst = eidx + EEDGE;

  // ---- param walk: JAX pytree leaves, dict keys sorted ----
  size_t po = 0;
  auto take = [&](size_t n){ const float* p = P + po; po += n; return p; };
  const float* at_emb = take(119*128);
  LinP at_gate; at_gate.b = take(128); at_gate.w = take(256*128);
  const float* at_ln_b = take(128); const float* at_ln_g = take(128);
  LinP at_meg;  at_meg.b  = take(128); at_meg.w  = take(16*128);
  LinP at_o1;   at_o1.b   = take(128); at_o1.w   = take(128*128);
  LinP at_o2;   at_o2.b   = take(128); at_o2.w   = take(128*128);
  struct { LinP a0,a1,g0,g1; const float *bn_b,*bn_g; } cp[3];
  for (int i = 0; i < 3; ++i){
    cp[i].a0.b = take(128); cp[i].a0.w = take(384*128);
    cp[i].a1.b = take(128); cp[i].a1.w = take(128*128);
    cp[i].bn_b = take(128); cp[i].bn_g = take(128);
    cp[i].g0.b = take(128); cp[i].g0.w = take(384*128);
    cp[i].g1.b = take(128); cp[i].g1.w = take(128*128);
  }
  LinP fc1; fc1.b = take(128); fc1.w = take(256*128);
  LinP fc2; fc2.b = take(1);   fc2.w = take(128);
  struct { LinP f0, f1; } fp[3];
  for (int i = 0; i < 3; ++i){
    fp[i].f0.b = take(256); fp[i].f0.w = take(128*256);
    fp[i].f1.b = take(128); fp[i].f1.w = take(256*128);
  }
  struct { const float *alpha,*b,*g; } ga[3], gf[3];
  for (int i = 0; i < 3; ++i){ ga[i].alpha = take(128); ga[i].b = take(128); ga[i].g = take(128); }
  for (int i = 0; i < 3; ++i){ gf[i].alpha = take(128); gf[i].b = take(128); gf[i].g = take(128); }
  struct {
    const float* beta_w; const float *bn_b,*bn_g;
    LinP e,k; const float *lna_b,*lna_g,*lnm_b,*lnm_g;
    LinP ml, mu0, mu1, q, skip, v;
  } mp[3];
  for (int i = 0; i < 3; ++i){
    mp[i].beta_w = take(384);
    mp[i].bn_b = take(128); mp[i].bn_g = take(128);
    mp[i].e.b = take(512); mp[i].e.w = take(128*512);
    mp[i].k.b = take(512); mp[i].k.w = take(128*512);
    mp[i].lna_b = take(384); mp[i].lna_g = take(384);
    mp[i].lnm_b = take(128); mp[i].lnm_g = take(128);
    mp[i].ml.b = take(128); mp[i].ml.w = take(384*128);
    mp[i].mu0.b = take(384); mp[i].mu0.w = take(384*384);
    mp[i].mu1.b = take(384); mp[i].mu1.w = take(384*384);
    mp[i].q.b = take(512); mp[i].q.w = take(128*512);
    mp[i].skip.b = take(128); mp[i].skip.w = take(128*128);
    mp[i].v.b = take(512); mp[i].v.w = take(128*512);
  }
  LinP rb1; rb1.b = take(128); rb1.w = take(128*128);
  LinP rb2; rb2.b = take(128); rb2.w = take(128*128);
  const float* s_b  = take(512);
  const float* s_wh = take(128*512);
  const float* s_wi = take(256*512);

  // ---- workspace bump allocator (16B aligned blocks) ----
  float* W0 = (float*)d_ws;
  size_t wo = 0;
  auto wal = [&](size_t nf){ wo = (wo + 3) & ~(size_t)3; float* p = W0 + wo; wo += nf; return p; };
  auto walh = [&](size_t nh){ return (_Float16*)wal((nh + 1) / 2); };

  float* hbuf   = wal((size_t)NN*128);
  float* efeat  = wal((size_t)EEDGE*128);
  float* dbuf   = wal(EEDGE);
  float* qbuf   = wal((size_t)NN*512);
  float* kbuf   = wal((size_t)NN*512);
  float* vbuf   = wal((size_t)NN*512);
  float* eebuf  = wal((size_t)EEDGE*512);
  float* T1     = wal((size_t)EEDGE*384);   // f32 alpha / rbf / ml-out (E x <=384)
  float* T3     = wal((size_t)EEDGE*128);   // f32 gate logits
  float* T4     = wal((size_t)EEDGE*384);   // f32 msg / mu1-out
  _Float16* T1h = walh((size_t)EEDGE*384);  // f16 cat / gated-msg
  _Float16* T2h = walh((size_t)EEDGE*384);  // f16 vcat / mlp mids
  _Float16* T3h = walh((size_t)EEDGE*384);  // f16 mu0 out
  _Float16* t256h = walh((size_t)NN*256);   // f16 ffn mid
  float* megpad = wal((size_t)NN*32);
  float* catN   = wal((size_t)NN*256);
  float* gbufN  = wal((size_t)NN*128);
  float* fusedN = wal((size_t)NN*128);
  float* tmpN   = wal((size_t)NN*128);
  float* aggN   = wal((size_t)NN*128);
  float* outN   = wal((size_t)NN*128);
  float* hlocal = wal((size_t)NN*128);
  float* hglob  = wal((size_t)NN*128);
  float* xrN    = wal((size_t)NN*128);
  float* tmpA   = wal((size_t)NN*128);
  float* tmpB   = wal((size_t)NN*128);
  float* hattn  = wal((size_t)NN*128);
  float* colm   = wal(128);
  float* colr   = wal(128);
  float* gsum   = wal((size_t)BGRAPH*128);
  float* gvar   = wal((size_t)BGRAPH*128);
  float* gcnt   = wal(BGRAPH);
  float* zB     = wal((size_t)BGRAPH*512);
  float* hB     = wal((size_t)BGRAPH*128);
  float* cB     = wal((size_t)BGRAPH*128);
  float* rB     = wal((size_t)BGRAPH*128);
  float* qstar  = wal((size_t)BGRAPH*256);
  float* engN   = wal(NN);
  float* aN     = wal(NN);
  float* mB     = wal(BGRAPH);
  float* sB     = wal(BGRAPH);
  float* fco    = wal((size_t)BGRAPH*128);
  _Float16* harena = walh((size_t)4*1024*1024);   // transposed f16 weights

  // ---- transpose/convert all GEMM weights to f16 [Nc][K] (L2-resident) ----
  _Float16* hp = harena;
  auto cvtW = [&](const float* w, int K, int Nc, int Kpad)->const _Float16*{
    _Float16* p = hp; size_t n = (size_t)Nc*Kpad; hp += n;
    k_wt<<<nb(n),256,0,stream>>>(w, p, K, Nc, Kpad, n);
    return p;
  };
  const _Float16* wt_atgate = cvtW(at_gate.w, 256, 128, 256);
  const _Float16* wt_atmeg  = cvtW(at_meg.w,   16, 128,  32);
  const _Float16* wt_ato1   = cvtW(at_o1.w,   128, 128, 128);
  const _Float16* wt_ato2   = cvtW(at_o2.w,   128, 128, 128);
  const _Float16* wt_rb1    = cvtW(rb1.w,     128, 128, 128);
  const _Float16* wt_rb2    = cvtW(rb2.w,     128, 128, 128);
  const _Float16 *wt_g0[3], *wt_g1[3], *wt_a0[3], *wt_a1[3];
  const _Float16 *wt_q[3], *wt_k[3], *wt_v[3], *wt_e[3];
  const _Float16 *wt_mu0[3], *wt_mu1[3], *wt_ml[3], *wt_skip[3];
  const _Float16 *wt_f0[3], *wt_f1[3];
  for (int l = 0; l < 3; ++l){
    wt_g0[l]  = cvtW(cp[l].g0.w, 384, 128, 384);
    wt_g1[l]  = cvtW(cp[l].g1.w, 128, 128, 128);
    wt_a0[l]  = cvtW(cp[l].a0.w, 384, 128, 384);
    wt_a1[l]  = cvtW(cp[l].a1.w, 128, 128, 128);
    wt_q[l]   = cvtW(mp[l].q.w,   128, 512, 128);
    wt_k[l]   = cvtW(mp[l].k.w,   128, 512, 128);
    wt_v[l]   = cvtW(mp[l].v.w,   128, 512, 128);
    wt_e[l]   = cvtW(mp[l].e.w,   128, 512, 128);
    wt_mu0[l] = cvtW(mp[l].mu0.w, 384, 384, 384);
    wt_mu1[l] = cvtW(mp[l].mu1.w, 384, 384, 384);
    wt_ml[l]  = cvtW(mp[l].ml.w,  384, 128, 384);
    wt_skip[l]= cvtW(mp[l].skip.w,128, 128, 128);
    wt_f0[l]  = cvtW(fp[l].f0.w,  128, 256, 128);
    wt_f1[l]  = cvtW(fp[l].f1.w,  256, 128, 256);
  }
  const _Float16* wt_fc1 = cvtW(fc1.w, 256, 128, 256);
  const _Float16* wt_wi  = cvtW(s_wi,  256, 512, 256);
  const _Float16* wt_wh  = cvtW(s_wh,  128, 512, 128);

  // ================= atom encoder =================
  k_emb<<<nb((size_t)NN*128),256,0,stream>>>(an, at_emb, catN, (size_t)NN*128);
  k_padmeg<<<nb((size_t)NN*32),256,0,stream>>>(meg_in, megpad, (size_t)NN*32);
  gemm32(stream, megpad, wt_atmeg, at_meg.b, catN+128, 256, NN, 32, 128, 0, false, false);
  gemm32(stream, catN, wt_atgate, at_gate.b, gbufN, 128, NN, 256, 128, 2, false, false);
  k_fuse<<<nb((size_t)NN*128),256,0,stream>>>(catN, gbufN, fusedN, (size_t)NN*128);
  gemm32(stream, fusedN, wt_ato1, at_o1.b, tmpN, 128, NN, 128, 128, 1, false, false);
  k_ln<<<NN,128,0,stream>>>(tmpN, at_ln_g, at_ln_b, 128);
  gemm32(stream, tmpN, wt_ato2, at_o2.b, hbuf, 128, NN, 128, 128, 0, false, false);

  // ================= RBF edge features =================
  k_dist<<<nb(EEDGE),256,0,stream>>>(evec, dbuf, EEDGE);
  k_rbf<<<nb((size_t)EEDGE*128),256,0,stream>>>(dbuf, T1, (size_t)EEDGE*128);
  gemm32(stream, T1, wt_rb1, rb1.b, T4, 128, EEDGE, 128, 128, 1, false, false);
  gemm32(stream, T4, wt_rb2, rb2.b, efeat, 128, EEDGE, 128, 128, 0, false, false);

  // ================= 3 layers =================
  for (int l = 0; l < 3; ++l){
    // ---- cart layer ----
    k_cart_cat<<<nb((size_t)EEDGE*384),256,0,stream>>>(hbuf, efeat, src, dst, T1h, (size_t)EEDGE*384);
    gemm16(stream, T1h, wt_g0[l], cp[l].g0.b, T2h, 128, EEDGE, 384, 128, 1, true);
    gemm16(stream, T2h, wt_g1[l], cp[l].g1.b, T3,  128, EEDGE, 128, 128, 0, false); // gate logits
    gemm16(stream, T1h, wt_a0[l], cp[l].a0.b, T2h, 128, EEDGE, 384, 128, 1, true);
    gemm16(stream, T2h, wt_a1[l], cp[l].a1.b, T4,  128, EEDGE, 128, 128, 0, false); // msg
    k_fill<<<nb((size_t)NN*128),256,0,stream>>>(aggN, 0.f, (size_t)NN*128);
    k_cart_scatter<<<nb((size_t)EEDGE*128),256,0,stream>>>(T3, T4, cdist, dst, aggN, (size_t)EEDGE*128);
    k_colstats<<<128,256,0,stream>>>(aggN, NN, colm, colr);
    k_bn_silu<true><<<nb((size_t)NN*128),256,0,stream>>>(aggN, colm, colr, cp[l].bn_g, cp[l].bn_b, hbuf, hlocal, (size_t)NN*128);

    // ---- matformer layer ----
    gemm32(stream, hbuf,  wt_q[l], mp[l].q.b, qbuf,  512, NN,    128, 512, 0, false, false);
    gemm32(stream, hbuf,  wt_k[l], mp[l].k.b, kbuf,  512, NN,    128, 512, 0, false, false);
    gemm32(stream, hbuf,  wt_v[l], mp[l].v.b, vbuf,  512, NN,    128, 512, 0, false, false);
    gemm32(stream, efeat, wt_e[l], mp[l].e.b, eebuf, 512, EEDGE, 128, 512, 0, false, false);
    k_fill<<<nb((size_t)NN*128),256,0,stream>>>(outN, 0.f, (size_t)NN*128);
    for (int hh = 0; hh < 4; ++hh){
      k_kcat_alpha<<<nb((size_t)EEDGE*384),256,0,stream>>>(kbuf, qbuf, eebuf, src, dst, hh, T1, (size_t)EEDGE*384);
      k_vcat<<<nb((size_t)EEDGE*384),256,0,stream>>>(vbuf, eebuf, src, dst, hh, T2h, (size_t)EEDGE*384);
      gemm16(stream, T2h, wt_mu0[l], mp[l].mu0.b, T3h, 384, EEDGE, 384, 384, 1, true);
      gemm16(stream, T3h, wt_mu1[l], mp[l].mu1.b, T4,  384, EEDGE, 384, 384, 0, false);
      k_ln_sigmul_h<<<EEDGE,128,0,stream>>>(T1, mp[l].lna_g, mp[l].lna_b, T4, T1h);
      gemm16(stream, T1h, wt_ml[l], mp[l].ml.b, T1, 128, EEDGE, 384, 128, 0, false);
      k_ln<<<EEDGE,128,0,stream>>>(T1, mp[l].lnm_g, mp[l].lnm_b, 128);
      k_mat_scatter<<<nb((size_t)EEDGE*128),256,0,stream>>>(T1, dst, outN, (size_t)EEDGE*128);
    }
    k_colstats<<<128,256,0,stream>>>(outN, NN, colm, colr);
    k_bn_silu<false><<<nb((size_t)NN*128),256,0,stream>>>(outN, colm, colr, mp[l].bn_g, mp[l].bn_b, nullptr, outN, (size_t)NN*128);
    gemm32(stream, hbuf, wt_skip[l], mp[l].skip.b, xrN, 128, NN, 128, 128, 0, false, false);
    k_beta<<<NN,128,0,stream>>>(outN, xrN, mp[l].beta_w, hglob);

    // ---- combine + graph norms + ffn ----
    k_add<<<nb((size_t)NN*128),256,0,stream>>>(hlocal, hglob, tmpA, (size_t)NN*128);
    k_fill<<<nb((size_t)BGRAPH*128),256,0,stream>>>(gsum, 0.f, (size_t)BGRAPH*128);
    k_fill<<<nb((size_t)BGRAPH*128),256,0,stream>>>(gvar, 0.f, (size_t)BGRAPH*128);
    k_fill<<<1,256,0,stream>>>(gcnt, 0.f, BGRAPH);
    k_gn_accum<<<nb((size_t)NN*128),256,0,stream>>>(tmpA, batch, gsum, gcnt, (size_t)NN*128);
    k_gn_center<<<nb((size_t)NN*128),256,0,stream>>>(tmpA, batch, gsum, gcnt, ga[l].alpha, tmpB, gvar, (size_t)NN*128);
    k_gn_final<<<nb((size_t)NN*128),256,0,stream>>>(tmpB, batch, gvar, gcnt, ga[l].g, ga[l].b, hbuf, hattn, (size_t)NN*128);
    gemm32(stream, hattn, wt_f0[l], fp[l].f0.b, t256h, 256, NN, 128, 256, 1, false, true); // silu->f16
    gemm16(stream, t256h, wt_f1[l], fp[l].f1.b, tmpA, 128, NN, 256, 128, 0, false);
    k_fill<<<nb((size_t)BGRAPH*128),256,0,stream>>>(gsum, 0.f, (size_t)BGRAPH*128);
    k_fill<<<nb((size_t)BGRAPH*128),256,0,stream>>>(gvar, 0.f, (size_t)BGRAPH*128);
    k_fill<<<1,256,0,stream>>>(gcnt, 0.f, BGRAPH);
    k_gn_accum<<<nb((size_t)NN*128),256,0,stream>>>(tmpA, batch, gsum, gcnt, (size_t)NN*128);
    k_gn_center<<<nb((size_t)NN*128),256,0,stream>>>(tmpA, batch, gsum, gcnt, gf[l].alpha, tmpB, gvar, (size_t)NN*128);
    k_gn_final<<<nb((size_t)NN*128),256,0,stream>>>(tmpB, batch, gvar, gcnt, gf[l].g, gf[l].b, hattn, hbuf, (size_t)NN*128);
  }

  // ================= set2set =================
  k_fill<<<nb((size_t)BGRAPH*256),256,0,stream>>>(qstar, 0.f, (size_t)BGRAPH*256);
  k_fill<<<nb((size_t)BGRAPH*128),256,0,stream>>>(hB, 0.f, (size_t)BGRAPH*128);
  k_fill<<<nb((size_t)BGRAPH*128),256,0,stream>>>(cB, 0.f, (size_t)BGRAPH*128);
  for (int step = 0; step < 3; ++step){
    gemm32(stream, qstar, wt_wi, s_b, zB, 512, BGRAPH, 256, 512, 0, false, false);
    gemm32(stream, hB, wt_wh, nullptr, zB, 512, BGRAPH, 128, 512, 0, true, false);  // accumulate
    k_lstm<<<nb((size_t)BGRAPH*128),256,0,stream>>>(zB, hB, cB, (size_t)BGRAPH*128);
    k_eng<<<NN,128,0,stream>>>(hbuf, hB, batch, engN);
    k_segmax<<<BGRAPH,256,0,stream>>>(engN, batch, mB, NN);
    k_expa<<<nb(NN),256,0,stream>>>(engN, batch, mB, aN, NN);
    k_segsum1<<<BGRAPH,256,0,stream>>>(aN, batch, sB, NN);
    k_rgather<<<BGRAPH,128,0,stream>>>(aN, hbuf, batch, sB, rB, NN);
    k_qstar<<<nb((size_t)BGRAPH*256),256,0,stream>>>(hB, rB, qstar, (size_t)BGRAPH*256);
  }
  gemm32(stream, qstar, wt_fc1, fc1.b, fco, 128, BGRAPH, 256, 128, 1, false, false);
  k_fc2k<<<BGRAPH,128,0,stream>>>(fco, fc2.w, fc2.b, (float*)d_out);
}
